// OverlappingLocalSelfAttention_10359461118550
// MI455X (gfx1250) — matrix operand order repaired and checked by reading, not compiled, not run on hardware
//
#include <hip/hip_runtime.h>
#include <hip/hip_bf16.h>

typedef __attribute__((ext_vector_type(16))) __bf16 v16bf;
typedef __attribute__((ext_vector_type(8)))  float  v8f;

union Frag {
    v16bf v;
    unsigned short u[16];
    unsigned int   w[8];
    uint4 q[2];
};

// hardware f32 -> bf16 (RTNE) via native type; lowers to v_cvt*bf16*
__device__ __forceinline__ unsigned short f2bf(float f) {
    union { __bf16 h; unsigned short u; } x;
    x.h = (__bf16)f;
    return x.u;
}
// packed pair: two converts fold into one v_cvt_pk_bf16_f32
__device__ __forceinline__ unsigned int pkbf(float a, float b) {
    union { struct { __bf16 l, h; } s; unsigned int u; } x;
    x.s.l = (__bf16)a;
    x.s.h = (__bf16)b;
    return x.u;
}

__device__ __forceinline__ v8f wmma_bf16(const Frag& a, const Frag& b, v8f c) {
    return __builtin_amdgcn_wmma_f32_16x16x32_bf16(false, a.v, false, b.v,
                                                   (short)0, c, false, false);
}

// gfx1250 async copy: global -> LDS, 128-bit per lane, tracked by ASYNCcnt
__device__ __forceinline__ void async_ld_b128(void* lds, const void* gp) {
    unsigned l = (unsigned)(unsigned long long)lds;   // LDS byte offset (addr[31:0])
    asm volatile("global_load_async_to_lds_b128 %0, %1, off"
                 :: "v"(l), "v"(gp) : "memory");
}
__device__ __forceinline__ void wait_async0() {
    asm volatile("s_wait_asynccnt 0" ::: "memory");
}

// ---------------------------------------------------------------------------
// Kernel 0: Wf (1024x1024 f32) -> bf16
// ---------------------------------------------------------------------------
__global__ __launch_bounds__(256) void cvt_wf(const float* __restrict__ W,
                                              unsigned short* __restrict__ Wb) {
    int i = blockIdx.x * 256 + threadIdx.x;
    Wb[i] = f2bf(W[i]);
}

// ---------------------------------------------------------------------------
// Kernel 1: QKV projection.  out[z][n][h][l][e] = sum_d X[n][l][h][d]*W[e][d]
// grid (32, 64, 3), block 256 (8 waves, 16 token-rows each => 128 rows/block)
// ---------------------------------------------------------------------------
__global__ __launch_bounds__(256) void qkv_proj(
    const float* __restrict__ vals, const float* __restrict__ keys,
    const float* __restrict__ qrys, const float* __restrict__ Wv,
    const float* __restrict__ Wk,   const float* __restrict__ Wq,
    unsigned short* __restrict__ qkv) {
    const int z = blockIdx.z;
    const float* src = (z == 0) ? vals : ((z == 1) ? keys : qrys);
    const float* W   = (z == 0) ? Wv   : ((z == 1) ? Wk   : Wq);
    const int nh = blockIdx.y;
    const int n = nh >> 4, h = nh & 15;
    const int rowBlock = blockIdx.x * 128;

    __shared__ unsigned short sW[64 * 72];     // [e][d], stride 72
    __shared__ unsigned short sOut[128 * 72];  // C staging, stride 72
    for (int i = threadIdx.x; i < 4096; i += 256)
        sW[(i >> 6) * 72 + (i & 63)] = f2bf(W[i]);
    __syncthreads();

    const int wave = threadIdx.x >> 5, lane = threadIdx.x & 31;
    const int m = lane & 15, kb = (lane >> 4) * 8;
    const int row = rowBlock + wave * 16 + m;
    const float* xp = src + ((size_t)n * 4096 + row) * 1024 + h * 64;

    Frag a[2];
#pragma unroll
    for (int kh = 0; kh < 2; ++kh)
#pragma unroll
        for (int t = 0; t < 8; ++t) {
            int g = t >> 2, tt = t & 3;
            int dd = kh * 32 + 2 * kb + 2 * t;
            a[kh].w[t] = pkbf(xp[dd], xp[dd + 1]);
        }

    const int rloc = wave * 16 + ((lane >> 4) << 3);
#pragma unroll
    for (int nt = 0; nt < 4; ++nt) {
        v8f c = {0.f, 0.f, 0.f, 0.f, 0.f, 0.f, 0.f, 0.f};
#pragma unroll
        for (int kh = 0; kh < 2; ++kh) {
            Frag b;
            const unsigned short* wp =
                &sW[(nt * 16 + (lane & 15)) * 72 + kh * 32 + ((lane >> 4) << 4)];
#pragma unroll
            for (int e = 0; e < 16; ++e) b.u[e] = wp[e];
            c = wmma_bf16(a[kh], b, c);
        }
        const int col = nt * 16 + (lane & 15);
#pragma unroll
        for (int r = 0; r < 8; ++r)
            sOut[(rloc + r) * 72 + col] = f2bf(c[r]);
    }
    __syncthreads();

    // vectorized writeback: 128 rows x 64 cols bf16 -> b128 stores
    const size_t obase = (((size_t)z * 4 + n) * 16 + h) * 4096 + rowBlock;
#pragma unroll
    for (int it = 0; it < 4; ++it) {
        int slot = it * 256 + threadIdx.x;
        int r = slot >> 3, seg = slot & 7;
        *(uint4*)(qkv + (obase + r) * 64 + seg * 8) =
            *(const uint4*)(sOut + r * 72 + seg * 8);
    }
}

// ---------------------------------------------------------------------------
// Kernel 2: overlapping-window attention with online softmax (64-col chunks).
// grid (nW=16, H=16, n=4), block 256 (8 waves x 32 query rows)
// ---------------------------------------------------------------------------
__global__ __launch_bounds__(256) void attn(
    const unsigned short* __restrict__ qkv, const int* __restrict__ mask,
    unsigned short* __restrict__ attout) {
    const int w = blockIdx.x, h = blockIdx.y, n = blockIdx.z;
    const int wave = threadIdx.x >> 5, lane = threadIdx.x & 31;

    // pool: [sK 4608][sVt 4608][sP 18432] u16 ; reused as [stage 18432] at end
    __shared__ unsigned short smem[27648];
    unsigned short* sK  = smem;           // [kcol][d], stride 72
    unsigned short* sVt = smem + 4608;    // [d][kcol], stride 72
    unsigned short* sP  = smem + 9216;    // [wave][mt][row16][64], stride 72

    const size_t planeV = (((size_t)0 * 4 + n) * 16 + h) * 4096;
    const size_t planeK = (((size_t)1 * 4 + n) * 16 + h) * 4096;
    const size_t planeQ = (((size_t)2 * 4 + n) * 16 + h) * 4096;

    const int m = lane & 15, kb = (lane >> 4) * 8;

    // Q fragments (A layout), 2 row-tiles x 2 K-halves
    Frag aq[2][2];
#pragma unroll
    for (int mt = 0; mt < 2; ++mt) {
        const unsigned short* qp =
            qkv + (planeQ + w * 256 + wave * 32 + mt * 16 + m) * 64;
#pragma unroll
        for (int kh = 0; kh < 2; ++kh)
#pragma unroll
            for (int t = 0; t < 8; ++t) {
                int g = t >> 2, tt = t & 3;
                int dd = kh * 32 + 2 * kb + 2 * t;
                aq[mt][kh].u[2 * t]     = qp[dd];
                aq[mt][kh].u[2 * t + 1] = qp[dd + 1];
            }
    }

    float qb[2][8];
#pragma unroll
    for (int mt = 0; mt < 2; ++mt)
#pragma unroll
        for (int r = 0; r < 8; ++r) {
            int qrow = w * 256 + wave * 32 + mt * 16 + r + ((lane >> 4) << 3);
            qb[mt][r] = mask[n * 4096 + qrow] ? 0.f : -1.25e9f;
        }

    float rmax[2][8], rsum[2][8];   // rsum: per-lane partial, reduced at end
    v8f o[2][4];
#pragma unroll
    for (int mt = 0; mt < 2; ++mt) {
#pragma unroll
        for (int r = 0; r < 8; ++r) { rmax[mt][r] = -__builtin_inff(); rsum[mt][r] = 0.f; }
#pragma unroll
        for (int nt = 0; nt < 4; ++nt)
            o[mt][nt] = (v8f){0.f, 0.f, 0.f, 0.f, 0.f, 0.f, 0.f, 0.f};
    }

    for (int ch = 0; ch < 5; ++ch) {          // 5 chunks x 64 key cols = 320
        __syncthreads();
        // stage K chunk: async global->LDS for valid rows, zero-fill edges
        for (int it = 0; it < 2; ++it) {
            int slot = it * 256 + threadIdx.x;
            int kr = slot >> 3, seg = slot & 7;
            int g = w * 256 - 32 + ch * 64 + kr;
            if (g >= 0 && g < 4096)
                async_ld_b128(sK + kr * 72 + seg * 8,
                              qkv + (planeK + g) * 64 + seg * 8);
            else
                *(uint4*)(sK + kr * 72 + seg * 8) = (uint4){0u, 0u, 0u, 0u};
        }
        // stage V chunk transposed (element scatter)
        for (int i = threadIdx.x; i < 4096; i += 256) {
            int kr = i >> 6, dc = i & 63;
            int g = w * 256 - 32 + ch * 64 + kr;
            unsigned short vv = 0;
            if (g >= 0 && g < 4096) vv = qkv[(planeV + g) * 64 + dc];
            sVt[dc * 72 + kr] = vv;
        }
        wait_async0();
        __syncthreads();

#pragma unroll
        for (int mt = 0; mt < 2; ++mt) {
            float sv[4][8];
#pragma unroll
            for (int kt = 0; kt < 4; ++kt) {
                v8f s = {0.f, 0.f, 0.f, 0.f, 0.f, 0.f, 0.f, 0.f};
#pragma unroll
                for (int kh = 0; kh < 2; ++kh) {
                    Frag b;
                    const unsigned short* kp =
                        sK + (kt * 16 + (lane & 15)) * 72 + kh * 32 + ((lane >> 4) << 4);
#pragma unroll
                    for (int e = 0; e < 16; ++e) b.u[e] = kp[e];
                    s = wmma_bf16(aq[mt][kh], b, s);
                }
                int gcol = w * 256 - 32 + ch * 64 + kt * 16 + (lane & 15);
                bool kvalid = (gcol >= 0) && (gcol < 4096);
#pragma unroll
                for (int r = 0; r < 8; ++r)
                    sv[kt][r] = kvalid ? (s[r] * 0.125f + qb[mt][r]) : -__builtin_inff();
            }
            // one online-softmax update per 64-col chunk
            unsigned short* pp =
                sP + ((wave * 2 + mt) * 16 + ((lane >> 4) << 3)) * 72 + (lane & 15);
#pragma unroll
            for (int r = 0; r < 8; ++r) {
                float tm = fmaxf(fmaxf(sv[0][r], sv[1][r]), fmaxf(sv[2][r], sv[3][r]));
                tm = fmaxf(tm, __shfl_xor(tm, 1, 16));
                tm = fmaxf(tm, __shfl_xor(tm, 2, 16));
                tm = fmaxf(tm, __shfl_xor(tm, 4, 16));
                tm = fmaxf(tm, __shfl_xor(tm, 8, 16));
                float om = rmax[mt][r];
                float nm = fmaxf(om, tm);
                float corr = (om == -__builtin_inff()) ? 0.f : __expf(om - nm);
                float p0 = (sv[0][r] == -__builtin_inff()) ? 0.f : __expf(sv[0][r] - nm);
                float p1 = (sv[1][r] == -__builtin_inff()) ? 0.f : __expf(sv[1][r] - nm);
                float p2 = (sv[2][r] == -__builtin_inff()) ? 0.f : __expf(sv[2][r] - nm);
                float p3 = (sv[3][r] == -__builtin_inff()) ? 0.f : __expf(sv[3][r] - nm);
                rsum[mt][r] = rsum[mt][r] * corr + ((p0 + p1) + (p2 + p3));
                rmax[mt][r] = nm;
#pragma unroll
                for (int nt = 0; nt < 4; ++nt) o[mt][nt][r] = o[mt][nt][r] * corr;
                pp[r * 72]      = f2bf(p0);
                pp[r * 72 + 16] = f2bf(p1);
                pp[r * 72 + 32] = f2bf(p2);
                pp[r * 72 + 48] = f2bf(p3);
            }
            // P (C-layout in LDS) -> A-fragments, O += P * V
            const unsigned short* pb = sP + (wave * 2 + mt) * 16 * 72 + m * 72;
#pragma unroll
            for (int pair = 0; pair < 2; ++pair) {
                Frag pa;
#pragma unroll
                for (int t = 0; t < 8; ++t) {
                    int g = t >> 2, tt = t & 3;
                    int kk = pair * 32 + 2 * kb + 2 * t;
                    pa.u[2 * t]     = pb[kk];
                    pa.u[2 * t + 1] = pb[kk + 1];
                }
#pragma unroll
                for (int nt = 0; nt < 4; ++nt) {
                    Frag b;
                    const unsigned short* vp =
                        sVt + (nt * 16 + (lane & 15)) * 72 + pair * 32 + ((lane >> 4) << 4);
#pragma unroll
                    for (int e = 0; e < 16; ++e) b.u[e] = vp[e];
                    o[mt][nt] = wmma_bf16(pa, b, o[mt][nt]);
                }
            }
        }
    }

    // finalize row sums (deferred cross-lane reduction), normalize, stage, store
#pragma unroll
    for (int mt = 0; mt < 2; ++mt)
#pragma unroll
        for (int r = 0; r < 8; ++r) {
            float s = rsum[mt][r];
            s += __shfl_xor(s, 1, 16);
            s += __shfl_xor(s, 2, 16);
            s += __shfl_xor(s, 4, 16);
            s += __shfl_xor(s, 8, 16);
            rsum[mt][r] = 1.f / s;
        }
    __syncthreads();
#pragma unroll
    for (int mt = 0; mt < 2; ++mt)
#pragma unroll
        for (int nt = 0; nt < 4; ++nt)
#pragma unroll
            for (int r = 0; r < 8; ++r) {
                int rloc = wave * 32 + mt * 16 + r + ((lane >> 4) << 3);
                int col = nt * 16 + (lane & 15);
                smem[rloc * 72 + col] = f2bf(o[mt][nt][r] * rsum[mt][r]);
            }
    __syncthreads();
#pragma unroll
    for (int it = 0; it < 8; ++it) {
        int slot = it * 256 + threadIdx.x;
        int r = slot >> 3, seg = slot & 7;
        *(uint4*)(attout + ((size_t)n * 4096 + w * 256 + r) * 1024 + h * 64 + seg * 8) =
            *(const uint4*)(smem + r * 72 + seg * 8);
    }
}

// ---------------------------------------------------------------------------
// Kernel 3: Y = X(16384x1024,bf16) * Wf^T + bf, f32 out.
// grid (128, 8), block 256; 128x128 output tile per block.
// ---------------------------------------------------------------------------
__global__ __launch_bounds__(256) void outproj(
    const unsigned short* __restrict__ X, const unsigned short* __restrict__ Wb,
    const float* __restrict__ bias, float* __restrict__ out) {
    const int mb = blockIdx.x * 128, nb = blockIdx.y * 128;
    __shared__ unsigned short sA[128 * 72];
    __shared__ unsigned short sB[128 * 72];
    const int wave = threadIdx.x >> 5, lane = threadIdx.x & 31;
    const int m = lane & 15, kb = (lane >> 4) * 8;

    v8f c[8];
#pragma unroll
    for (int nt = 0; nt < 8; ++nt) c[nt] = (v8f){0.f, 0.f, 0.f, 0.f, 0.f, 0.f, 0.f, 0.f};

    for (int kc = 0; kc < 1024; kc += 64) {
        __syncthreads();
        for (int it = 0; it < 4; ++it) {
            int slot = it * 256 + threadIdx.x;
            int r = slot >> 3, seg = slot & 7;
            const unsigned short* xs = X + ((size_t)(mb + r)) * 1024 + kc + seg * 8;
            const unsigned short* ws = Wb + ((size_t)(nb + r)) * 1024 + kc + seg * 8;
            async_ld_b128(sA + r * 72 + seg * 8, xs);   // async global->LDS
            async_ld_b128(sB + r * 72 + seg * 8, ws);
            if (kc + 64 < 1024) {
                __builtin_prefetch(xs + 64, 0, 1);      // next K-chunk
                __builtin_prefetch(ws + 64, 0, 1);
            }
        }
        wait_async0();
        __syncthreads();

        Frag a[2];
#pragma unroll
        for (int kh = 0; kh < 2; ++kh)
#pragma unroll
            for (int t = 0; t < 8; ++t) {
                int g = t >> 2, tt = t & 3;
                int kk = kh * 32 + 2 * kb + 2 * t;
                a[kh].u[2 * t]     = sA[(wave * 16 + m) * 72 + kk];
                a[kh].u[2 * t + 1] = sA[(wave * 16 + m) * 72 + kk + 1];
            }
#pragma unroll
        for (int nt = 0; nt < 8; ++nt)
#pragma unroll
            for (int kh = 0; kh < 2; ++kh) {
                Frag b;
                const unsigned short* wp =
                    sB + (nt * 16 + (lane & 15)) * 72 + kh * 32 + ((lane >> 4) << 4);
#pragma unroll
                for (int e = 0; e < 16; ++e) b.u[e] = wp[e];
                c[nt] = wmma_bf16(a[kh], b, c[nt]);
            }
    }

#pragma unroll
    for (int nt = 0; nt < 8; ++nt) {
        int col = nb + nt * 16 + (lane & 15);
        float bv = bias[col];
#pragma unroll
        for (int r = 0; r < 8; ++r) {
            int row = mb + wave * 16 + r + ((lane >> 4) << 3);
            out[(size_t)row * 1024 + col] = c[nt][r] + bv;
        }
    }
}

// ---------------------------------------------------------------------------
extern "C" void kernel_launch(void* const* d_in, const int* in_sizes, int n_in,
                              void* d_out, int out_size, void* d_ws, size_t ws_size,
                              hipStream_t stream) {
    (void)in_sizes; (void)n_in; (void)out_size; (void)ws_size;
    const float* vals = (const float*)d_in[0];
    const float* keys = (const float*)d_in[1];
    const float* qrys = (const float*)d_in[2];
    const int*   msk  = (const int*)d_in[3];
    const float* Wv   = (const float*)d_in[4];
    const float* Wk   = (const float*)d_in[5];
    const float* Wq   = (const float*)d_in[6];
    const float* Wf   = (const float*)d_in[7];
    const float* bf   = (const float*)d_in[8];

    unsigned short* qkv = (unsigned short*)d_ws;                   // 3*4*16*4096*64
    unsigned short* att = qkv + (size_t)3 * 4 * 16 * 4096 * 64;    // 4*4096*1024
    unsigned short* wfb = att + (size_t)4 * 4096 * 1024;           // 1024*1024
    float* out = (float*)d_out;

    cvt_wf<<<4096, 256, 0, stream>>>(Wf, wfb);
    qkv_proj<<<dim3(32, 64, 3), 256, 0, stream>>>(vals, keys, qrys, Wv, Wk, Wq, qkv);
    attn<<<dim3(16, 16, 4), 256, 0, stream>>>(qkv, msk, att);
    outproj<<<dim3(128, 8), 256, 0, stream>>>(att, wfb, bf, out);
}
